// MixtureOfBidders_89627377533439
// MI455X (gfx1250) — matrix operand order repaired and checked by reading, not compiled, not run on hardware
//
#include <hip/hip_runtime.h>
#include <hip/hip_bf16.h>

// Problem constants (from reference): B=2, S=1024 -> T=2048 tokens
#define T_TOKENS 2048
#define H_DIM    1024
#define I_DIM    4096
#define E_NUM    8
#define CAP      2048   // worst-case tokens per expert

typedef __attribute__((ext_vector_type(16))) __bf16         v16bf;
typedef __attribute__((ext_vector_type(8)))  float          v8f;
typedef __attribute__((ext_vector_type(4)))  unsigned int   u32x4;
typedef __attribute__((ext_vector_type(4)))  float          f32x4;
typedef __attribute__((ext_vector_type(8)))  unsigned short u16x8;

union FragB16 {
    v16bf v;
    unsigned short h[16];
    u32x4 q[2];
};

// Branchless round-to-nearest-even f32 -> bf16 (no NaN special case: straight-line VALU)
__device__ __forceinline__ unsigned short f32_to_bf16_rte(float f) {
    union { float f; unsigned int u; } c; c.f = f;
    unsigned int u = c.u + (0x7fffu + ((c.u >> 16) & 1u));
    return (unsigned short)(u >> 16);
}

// ---------------------------------------------------------------------------
// 0) zero output + per-expert counters
// ---------------------------------------------------------------------------
__global__ void moe_zero_kernel(float* __restrict__ out, int n, int* __restrict__ cnt) {
    int i = blockIdx.x * blockDim.x + threadIdx.x;
    if (i < n) out[i] = 0.0f;
    if (i < E_NUM) cnt[i] = 0;
}

// ---------------------------------------------------------------------------
// 1) cast hidden states fp32 -> bf16
// ---------------------------------------------------------------------------
__global__ void moe_cast_x_kernel(const float* __restrict__ x,
                                  unsigned short* __restrict__ xb, int n) {
    int i = blockIdx.x * blockDim.x + threadIdx.x;
    if (i < n) xb[i] = f32_to_bf16_rte(x[i]);
}

// ---------------------------------------------------------------------------
// 2) router: one wave32 per token
// ---------------------------------------------------------------------------
__global__ void moe_router_kernel(const float* __restrict__ x,
                                  const float* __restrict__ cw,
                                  const float* __restrict__ cb,
                                  const float* __restrict__ wealth,
                                  int* __restrict__ cnt,
                                  int* __restrict__ idx_list,
                                  float* __restrict__ wt_list) {
    int t    = (blockIdx.x * blockDim.x + threadIdx.x) >> 5;
    int lane = threadIdx.x & 31;
    if (t >= T_TOKENS) return;

    float acc[E_NUM];
#pragma unroll
    for (int e = 0; e < E_NUM; ++e) acc[e] = 0.0f;

    const float* xr = x + (size_t)t * H_DIM;
    for (int h = lane; h < H_DIM; h += 32) {
        float xv = xr[h];
#pragma unroll
        for (int e = 0; e < E_NUM; ++e) acc[e] += xv * cw[e * H_DIM + h];
    }
#pragma unroll
    for (int e = 0; e < E_NUM; ++e) {
#pragma unroll
        for (int off = 16; off > 0; off >>= 1)
            acc[e] += __shfl_xor(acc[e], off, 32);
    }

    if (lane == 0) {
        float bid[E_NUM];
#pragma unroll
        for (int e = 0; e < E_NUM; ++e) {
            float z = acc[e] + cb[e];
            bid[e] = (1.0f / (1.0f + __expf(-z))) * wealth[e];
        }
        // top-2, lowest index wins ties (matches jax top_k)
        int e0 = 0; float b0 = bid[0];
#pragma unroll
        for (int e = 1; e < E_NUM; ++e) if (bid[e] > b0) { b0 = bid[e]; e0 = e; }
        int e1 = (e0 == 0) ? 1 : 0; float b1 = bid[e1];
#pragma unroll
        for (int e = 0; e < E_NUM; ++e)
            if (e != e0 && e != e1 && bid[e] > b1) { b1 = bid[e]; e1 = e; }
        float ex = __expf(b1 - b0);     // b0 >= b1
        float w0 = 1.0f / (1.0f + ex);
        float w1 = ex * w0;

        int s0 = atomicAdd(&cnt[e0], 1);
        idx_list[e0 * CAP + s0] = t;
        wt_list [e0 * CAP + s0] = w0;
        int s1 = atomicAdd(&cnt[e1], 1);
        idx_list[e1 * CAP + s1] = t;
        wt_list [e1 * CAP + s1] = w1;
    }
}

// ---------------------------------------------------------------------------
// 3) exclusive scan of 8 counters -> compact row offsets
// ---------------------------------------------------------------------------
__global__ void moe_scan_kernel(const int* __restrict__ cnt, int* __restrict__ offs) {
    if (threadIdx.x == 0 && blockIdx.x == 0) {
        int a = 0;
        for (int e = 0; e < E_NUM; ++e) { offs[e] = a; a += cnt[e]; }
    }
}

// ---------------------------------------------------------------------------
// 4) stage A: grouped GEMM  A = silu(X*Wg^T) * (X*Wu^T)  -> bf16 Abuf
//    block tile 64(M) x 64(N of I); 8 waves = 4(M) x 2(N); each wave 16x32
//    (one A-frag reused across two B-frags per matrix -> 4 wmma / 10 ds_loads)
// ---------------------------------------------------------------------------
__global__ __launch_bounds__(256)
void moe_stage_a_kernel(const unsigned short* __restrict__ xb,
                        const float* __restrict__ gate_w,
                        const float* __restrict__ up_w,
                        const int* __restrict__ cnt,
                        const int* __restrict__ offs,
                        const int* __restrict__ idx_list,
                        unsigned short* __restrict__ abuf) {
    const int e  = blockIdx.z;
    const int m0 = blockIdx.y * 64;
    const int n0 = blockIdx.x * 64;
    const int count = cnt[e];
    if (m0 >= count) return;
    const int base = offs[e];

    __shared__ unsigned short Xs[64 * 32];   // [row][k]
    __shared__ unsigned short Gs[64 * 32];   // [n][k]
    __shared__ unsigned short Us[64 * 32];

    const int tid  = threadIdx.x;
    const int lane = tid & 31;
    const int wid  = tid >> 5;
    const int wm   = wid >> 1;     // 0..3 : 16-row slab
    const int wn   = wid & 1;      // 0..1 : 32-col slab
    const int half = lane >> 4;
    const int l15  = lane & 15;

    // X loader: row = tid/4 (0..63), 8 bf16 at col (tid&3)*8  (16B)
    const int rL = tid >> 2;
    const int cL = (tid & 3) * 8;
    int slotX = m0 + rL; if (slotX >= count) slotX = count - 1;
    const unsigned short* xrow = xb + (size_t)idx_list[e * CAP + slotX] * H_DIM;

    const float* grow = gate_w + (size_t)e * I_DIM * H_DIM + (size_t)(n0 + rL) * H_DIM;
    const float* urow = up_w   + (size_t)e * I_DIM * H_DIM + (size_t)(n0 + rL) * H_DIM;

    v8f accg[2] = {{}, {}};
    v8f accu[2] = {{}, {}};

    for (int kc = 0; kc < H_DIM; kc += 32) {
        // --- fill LDS ---
        *(u16x8*)(&Xs[rL * 32 + cL]) = *(const u16x8*)(xrow + kc + cL);
        {
            f32x4 a0 = *(const f32x4*)(grow + kc + cL);
            f32x4 a1 = *(const f32x4*)(grow + kc + cL + 4);
            f32x4 b0 = *(const f32x4*)(urow + kc + cL);
            f32x4 b1 = *(const f32x4*)(urow + kc + cL + 4);
            u16x8 og, ou;
#pragma unroll
            for (int j = 0; j < 4; ++j) {
                og[j]     = f32_to_bf16_rte(a0[j]);
                og[4 + j] = f32_to_bf16_rte(a1[j]);
                ou[j]     = f32_to_bf16_rte(b0[j]);
                ou[4 + j] = f32_to_bf16_rte(b1[j]);
            }
            *(u16x8*)(&Gs[rL * 32 + cL]) = og;
            *(u16x8*)(&Us[rL * 32 + cL]) = ou;
        }
        if (kc + 32 < H_DIM) {   // stream hint: global_prefetch_b8
            __builtin_prefetch(grow + kc + 32 + cL, 0, 1);
            __builtin_prefetch(urow + kc + 32 + cL, 0, 1);
        }
        __syncthreads();

        // --- fragments (contiguous K = half*16 + s, same order in A and B) ---
        FragB16 fa, fg0, fg1, fu0, fu1;
        {
            const unsigned short* p = &Xs[(wm * 16 + l15) * 32 + half * 16];
            fa.q[0] = *(const u32x4*)p;          // K = half*16 .. +7
            fa.q[1] = *(const u32x4*)(p + 8);    // K = half*16+8 .. +15
        }
        {
            const unsigned short* p0 = &Gs[(wn * 32 + l15) * 32 + half * 16];
            fg0.q[0] = *(const u32x4*)p0;        // K = half*16 .. +7
            fg0.q[1] = *(const u32x4*)(p0 + 8);  // K = half*16+8 .. +15
            const unsigned short* p1 = &Gs[(wn * 32 + 16 + l15) * 32 + half * 16];
            fg1.q[0] = *(const u32x4*)p1;
            fg1.q[1] = *(const u32x4*)(p1 + 8);
            const unsigned short* q0 = &Us[(wn * 32 + l15) * 32 + half * 16];
            fu0.q[0] = *(const u32x4*)q0;
            fu0.q[1] = *(const u32x4*)(q0 + 8);
            const unsigned short* q1 = &Us[(wn * 32 + 16 + l15) * 32 + half * 16];
            fu1.q[0] = *(const u32x4*)q1;
            fu1.q[1] = *(const u32x4*)(q1 + 8);
        }
        accg[0] = __builtin_amdgcn_wmma_f32_16x16x32_bf16(false, fa.v, false, fg0.v,
                                                          (short)0, accg[0], false, false);
        accg[1] = __builtin_amdgcn_wmma_f32_16x16x32_bf16(false, fa.v, false, fg1.v,
                                                          (short)0, accg[1], false, false);
        accu[0] = __builtin_amdgcn_wmma_f32_16x16x32_bf16(false, fa.v, false, fu0.v,
                                                          (short)0, accu[0], false, false);
        accu[1] = __builtin_amdgcn_wmma_f32_16x16x32_bf16(false, fa.v, false, fu1.v,
                                                          (short)0, accu[1], false, false);
        __syncthreads();
    }

    // epilogue: silu(g)*u -> bf16 Abuf
#pragma unroll
    for (int t = 0; t < 2; ++t) {
        const int nglob = n0 + wn * 32 + t * 16 + l15;
#pragma unroll
        for (int r = 0; r < 8; ++r) {
            int mrow = wm * 16 + half * 8 + r;   // C/D layout: M = r + 8*half
            int slot = m0 + mrow;
            if (slot < count) {
                float g = accg[t][r], u = accu[t][r];
                float a = (g / (1.0f + __expf(-g))) * u;
                abuf[(size_t)(base + slot) * I_DIM + nglob] = f32_to_bf16_rte(a);
            }
        }
    }
}

// ---------------------------------------------------------------------------
// 5) stage B: out += wt * (Abuf * Wd^T)   (atomicAdd f32, top-2 accumulation)
//    same 64x64 tiling; N over H
// ---------------------------------------------------------------------------
__global__ __launch_bounds__(256)
void moe_stage_b_kernel(const unsigned short* __restrict__ abuf,
                        const float* __restrict__ down_w,
                        const int* __restrict__ cnt,
                        const int* __restrict__ offs,
                        const int* __restrict__ idx_list,
                        const float* __restrict__ wt_list,
                        float* __restrict__ out) {
    const int e  = blockIdx.z;
    const int m0 = blockIdx.y * 64;
    const int n0 = blockIdx.x * 64;     // over H
    const int count = cnt[e];
    if (m0 >= count) return;
    const int base = offs[e];

    __shared__ unsigned short As[64 * 32];
    __shared__ unsigned short Ds[64 * 32];

    const int tid  = threadIdx.x;
    const int lane = tid & 31;
    const int wid  = tid >> 5;
    const int wm   = wid >> 1;
    const int wn   = wid & 1;
    const int half = lane >> 4;
    const int l15  = lane & 15;

    const int rL = tid >> 2;
    const int cL = (tid & 3) * 8;
    int slotX = m0 + rL; if (slotX >= count) slotX = count - 1;
    const unsigned short* arow = abuf + (size_t)(base + slotX) * I_DIM;

    const float* drow = down_w + (size_t)e * H_DIM * I_DIM + (size_t)(n0 + rL) * I_DIM;

    v8f acc[2] = {{}, {}};

    for (int kc = 0; kc < I_DIM; kc += 32) {
        *(u16x8*)(&As[rL * 32 + cL]) = *(const u16x8*)(arow + kc + cL);
        {
            f32x4 a0 = *(const f32x4*)(drow + kc + cL);
            f32x4 a1 = *(const f32x4*)(drow + kc + cL + 4);
            u16x8 od;
#pragma unroll
            for (int j = 0; j < 4; ++j) {
                od[j]     = f32_to_bf16_rte(a0[j]);
                od[4 + j] = f32_to_bf16_rte(a1[j]);
            }
            *(u16x8*)(&Ds[rL * 32 + cL]) = od;
        }
        if (kc + 32 < I_DIM)
            __builtin_prefetch(drow + kc + 32 + cL, 0, 1);
        __syncthreads();

        FragB16 fa, fd0, fd1;
        {
            const unsigned short* p = &As[(wm * 16 + l15) * 32 + half * 16];
            fa.q[0] = *(const u32x4*)p;
            fa.q[1] = *(const u32x4*)(p + 8);
        }
        {
            const unsigned short* p0 = &Ds[(wn * 32 + l15) * 32 + half * 16];
            fd0.q[0] = *(const u32x4*)p0;
            fd0.q[1] = *(const u32x4*)(p0 + 8);
            const unsigned short* p1 = &Ds[(wn * 32 + 16 + l15) * 32 + half * 16];
            fd1.q[0] = *(const u32x4*)p1;
            fd1.q[1] = *(const u32x4*)(p1 + 8);
        }
        acc[0] = __builtin_amdgcn_wmma_f32_16x16x32_bf16(false, fa.v, false, fd0.v,
                                                         (short)0, acc[0], false, false);
        acc[1] = __builtin_amdgcn_wmma_f32_16x16x32_bf16(false, fa.v, false, fd1.v,
                                                         (short)0, acc[1], false, false);
        __syncthreads();
    }

#pragma unroll
    for (int t = 0; t < 2; ++t) {
        const int hglob = n0 + wn * 32 + t * 16 + l15;
#pragma unroll
        for (int r = 0; r < 8; ++r) {
            int mrow = wm * 16 + half * 8 + r;
            int slot = m0 + mrow;
            if (slot < count) {
                int   tok = idx_list[e * CAP + slot];
                float w   = wt_list [e * CAP + slot];
                atomicAdd(&out[(size_t)tok * H_DIM + hglob], acc[t][r] * w);
            }
        }
    }
}

// ---------------------------------------------------------------------------
// launch
// ---------------------------------------------------------------------------
extern "C" void kernel_launch(void* const* d_in, const int* in_sizes, int n_in,
                              void* d_out, int out_size, void* d_ws, size_t ws_size,
                              hipStream_t stream) {
    (void)in_sizes; (void)n_in; (void)out_size; (void)ws_size;
    const float* x      = (const float*)d_in[0];   // (B,S,H)
    const float* conf_w = (const float*)d_in[1];   // (E,H)
    const float* conf_b = (const float*)d_in[2];   // (E,)
    const float* gate_w = (const float*)d_in[3];   // (E,I,H)
    const float* up_w   = (const float*)d_in[4];   // (E,I,H)
    const float* down_w = (const float*)d_in[5];   // (E,H,I)
    const float* wealth = (const float*)d_in[6];   // (E,)
    float* out = (float*)d_out;                    // (B,S,H) fp32

    // workspace layout (bytes)
    char* ws = (char*)d_ws;
    int*   cnt  = (int*)  (ws + 0);                         // 8 ints
    int*   offs = (int*)  (ws + 128);                       // 8 ints
    int*   idx  = (int*)  (ws + 256);                       // E*CAP ints   (64 KB)
    float* wts  = (float*)(ws + 256 + 65536);               // E*CAP floats (64 KB)
    unsigned short* xb   = (unsigned short*)(ws + 131328);  // T*H bf16     (4 MB)
    unsigned short* abuf = (unsigned short*)(ws + 4325632); // 4096*I bf16  (32 MB)

    const int outN = T_TOKENS * H_DIM;

    moe_zero_kernel  <<<(outN + 255) / 256, 256, 0, stream>>>(out, outN, cnt);
    moe_cast_x_kernel<<<(outN + 255) / 256, 256, 0, stream>>>(x, xb, outN);
    moe_router_kernel<<<T_TOKENS / 8, 256, 0, stream>>>(x, conf_w, conf_b, wealth,
                                                        cnt, idx, wts);
    moe_scan_kernel  <<<1, 32, 0, stream>>>(cnt, offs);

    dim3 ga(I_DIM / 64, CAP / 64, E_NUM);
    moe_stage_a_kernel<<<ga, 256, 0, stream>>>(xb, gate_w, up_w, cnt, offs, idx, abuf);

    dim3 gb(H_DIM / 64, CAP / 64, E_NUM);
    moe_stage_b_kernel<<<gb, 256, 0, stream>>>(abuf, down_w, cnt, offs, idx, wts, out);
}
